// RNN_47949014892907
// MI455X (gfx1250) — matrix operand order repaired and checked by reading, not compiled, not run on hardware
//
#include <hip/hip_runtime.h>
#include <math.h>

// ---------------- dims ----------------
#define EMB    1024
#define HID    1024
#define SEQ    128
#define BATCH  128
#define VOCAB  10000
#define SB     (SEQ * BATCH)   // 16384

// ---------------- WMMA types ----------------
typedef __attribute__((ext_vector_type(16))) __bf16 v16bf;
typedef __attribute__((ext_vector_type(8)))  __bf16 v8bf;
typedef __attribute__((ext_vector_type(8)))  float  v8f;

// 16-byte POD for global->LDS staging copies
struct alignas(16) U4 { unsigned int a, b, c, d; };

__device__ __forceinline__ v8f wmma_bf16(v16bf a, v16bf b, v8f c) {
    return __builtin_amdgcn_wmma_f32_16x16x32_bf16(
        false, a, false, b, (short)0, c, false, false);
}

__device__ __forceinline__ unsigned short f2bf(float f) {
    unsigned int u = __float_as_uint(f);
    u += 0x7FFFu + ((u >> 16) & 1u);
    return (unsigned short)(u >> 16);
}

__device__ __forceinline__ v16bf make_frag(v8bf lo, v8bf hi) {
    return __builtin_shufflevector(lo, hi, 0,1,2,3,4,5,6,7,8,9,10,11,12,13,14,15);
}

// LDS row stride: 32 bf16 (64B) data + 8 bf16 (16B) pad = 40 ushorts (80B) -> bank-conflict free
#define LROW 40

// ------------- LDS-pipelined accumulate: acc[TN] += A[16 x K] @ Wrows[TN*16 x K]^T -------------
// lds: 2 * TN*16*LROW ushorts (double buffered).  All 8 waves share the staged B chunk.
template<int TN>
__device__ __forceinline__ void accum_lds(v8f* acc, unsigned short* lds,
                                          const unsigned short* __restrict__ abase,
                                          const unsigned short* __restrict__ wrows,
                                          int K, int tid) {
    const int lane = tid & 31;
    const int nloc = lane & 15;
    const int half = lane >> 4;
    const int row0 = tid >> 2;        // 0..63
    const int q0   = tid & 3;         // 16B segment within 64B chunk row
    const int BUF  = TN * 16 * LROW;  // ushorts per LDS buffer

    // ---- prologue: stage chunk 0 (global -> LDS) + prefetch A chunk 0 ----
    {
        U4 g0 = *(const U4*)(wrows + (size_t)row0 * K + q0 * 8);
        *(U4*)&lds[row0 * LROW + q0 * 8] = g0;
        if (TN == 5 && tid < 64) {     // rows 64..79 (wave-uniform: waves 0..1)
            U4 g1 = *(const U4*)(wrows + (size_t)(64 + row0) * K + q0 * 8);
            *(U4*)&lds[(64 + row0) * LROW + q0 * 8] = g1;
        }
    }
    v8bf alo = *(const v8bf*)(const void*)(abase);
    v8bf ahi = *(const v8bf*)(const void*)(abase + 16);
    __syncthreads();

    const int nk = K / 32;
    int curoff = 0;                    // toggled LDS buffer offset (ushorts)
#pragma unroll 1
    for (int c = 0; c < nk - 1; ++c) {
        const int k = c * 32;

        // 1) prefetch chunk c+1 (global -> regs); s_wait lands after the WMMAs below
        U4 g0 = *(const U4*)(wrows + (size_t)row0 * K + k + 32 + q0 * 8);
        U4 g1;
        if (TN == 5 && tid < 64)
            g1 = *(const U4*)(wrows + (size_t)(64 + row0) * K + k + 32 + q0 * 8);
        v8bf alo_n = *(const v8bf*)(const void*)(abase + k + 32);
        v8bf ahi_n = *(const v8bf*)(const void*)(abase + k + 48);

        // 2) load ALL B fragments first (distinct regs -> partial dscnt waits), then WMMAs
        const unsigned short* bbase = &lds[curoff + nloc * LROW + half * 8];
        v16bf bf[TN];
#pragma unroll
        for (int j = 0; j < TN; ++j) {
            const unsigned short* p = bbase + j * (16 * LROW);
            v8bf blo = *(const v8bf*)(const void*)(p);
            v8bf bhi = *(const v8bf*)(const void*)(p + 16);
            bf[j] = make_frag(blo, bhi);
        }
        v16bf a = make_frag(alo, ahi);
#pragma unroll
        for (int j = 0; j < TN; ++j)
            acc[j] = wmma_bf16(a, bf[j], acc[j]);

        // 3) store staged chunk c+1 into the other LDS buffer
        const int nxt = BUF - curoff;
        *(U4*)&lds[nxt + row0 * LROW + q0 * 8] = g0;
        if (TN == 5 && tid < 64)
            *(U4*)&lds[nxt + (64 + row0) * LROW + q0 * 8] = g1;
        curoff = nxt;
        alo = alo_n; ahi = ahi_n;
        __syncthreads();
    }

    // ---- last chunk ----
    {
        const unsigned short* bbase = &lds[curoff + nloc * LROW + half * 8];
        v16bf bf[TN];
#pragma unroll
        for (int j = 0; j < TN; ++j) {
            const unsigned short* p = bbase + j * (16 * LROW);
            v8bf blo = *(const v8bf*)(const void*)(p);
            v8bf bhi = *(const v8bf*)(const void*)(p + 16);
            bf[j] = make_frag(blo, bhi);
        }
        v16bf a = make_frag(alo, ahi);
#pragma unroll
        for (int j = 0; j < TN; ++j)
            acc[j] = wmma_bf16(a, bf[j], acc[j]);
    }
    __syncthreads();   // make LDS safe for a following pass
}

// ---------------- helper kernels ----------------
__global__ __launch_bounds__(256) void to_bf16_kernel(const float* __restrict__ src,
                                                      unsigned short* __restrict__ dst, int n) {
    int i = blockIdx.x * 256 + threadIdx.x;
    if (i < n) dst[i] = f2bf(src[i]);
}

__global__ __launch_bounds__(256) void gather_embed_kernel(const int* __restrict__ ids,
                                                           const float* __restrict__ emb,
                                                           unsigned short* __restrict__ xe) {
    const int row = ids[blockIdx.x];
    const float* src = emb + (size_t)row * EMB;
    unsigned short* dst = xe + (size_t)blockIdx.x * EMB;
    for (int j = threadIdx.x; j < EMB; j += 256)
        dst[j] = f2bf(src[j]);
}

// ---------------- generic bf16 WMMA GEMM with LDS-staged B ----------------
// out[M,N] (f32) = A[M,K]_bf16 @ W[N,K]_bf16^T (+ bias)
// grid: x = Mtiles/8, y = Ntiles/TN (exact); block = 256 (8 waves, wave = M-tile, shared N-group)
template<int TN, bool HAS_BIAS>
__global__ __launch_bounds__(256) void gemm_wmma_bf16(const unsigned short* __restrict__ A,
                                                      const unsigned short* __restrict__ W,
                                                      const float* __restrict__ bias,
                                                      float* __restrict__ out,
                                                      int K, int N) {
    __shared__ __align__(16) unsigned short ldsB[2 * TN * 16 * LROW];
    const int tid   = threadIdx.x;
    const int lane  = tid & 31;
    const int wave  = tid >> 5;
    const int mtile = blockIdx.x * 8 + wave;
    const int nt0   = blockIdx.y * TN;
    const int nloc  = lane & 15;
    const int half  = lane >> 4;

    const v8f z = {0.f,0.f,0.f,0.f,0.f,0.f,0.f,0.f};
    v8f acc[TN];
#pragma unroll
    for (int j = 0; j < TN; ++j) acc[j] = z;

    accum_lds<TN>(acc, ldsB,
                  A + ((size_t)mtile * 16 + nloc) * K + half * 8,
                  W + (size_t)nt0 * 16 * K,
                  K, tid);

#pragma unroll
    for (int j = 0; j < TN; ++j) {
        const int n  = (nt0 + j) * 16 + nloc;
        float bv = 0.0f;
        if (HAS_BIAS) bv = bias[n];
#pragma unroll
        for (int r = 0; r < 8; ++r) {
            const int m = mtile * 16 + half * 8 + r;
            out[(size_t)m * N + n] = acc[j][r] + bv;
        }
    }
}

// ---------------- recurrent step kernels ----------------
// Block = one N-group (blockIdx.x = ngrp 0..15), wave = M-tile 0..7; waves share staged W rows.
// Layer 0: h0' = tanh(h0 @ wh0^T + x0_t + b0).  grid 16 x 256.
__global__ __launch_bounds__(256) void rnn_step_l0(const unsigned short* __restrict__ h_in,
                                                   const unsigned short* __restrict__ wh,
                                                   const float* __restrict__ x0_t,
                                                   const float* __restrict__ bias,
                                                   unsigned short* __restrict__ h_out,
                                                   float* __restrict__ h_final) {
    __shared__ __align__(16) unsigned short ldsB[2 * 4 * 16 * LROW];
    const int tid   = threadIdx.x;
    const int lane  = tid & 31;
    const int mtile = tid >> 5;            // 0..7  (BATCH/16)
    const int ngrp  = blockIdx.x;          // 0..15 (HID/64)
    const int nloc  = lane & 15;
    const int half  = lane >> 4;

    const v8f z = {0.f,0.f,0.f,0.f,0.f,0.f,0.f,0.f};
    v8f acc[4] = {z, z, z, z};

    accum_lds<4>(acc, ldsB,
                 h_in + ((size_t)mtile * 16 + nloc) * HID + half * 8,
                 wh   + (size_t)ngrp * 64 * HID,
                 HID, tid);

#pragma unroll
    for (int j = 0; j < 4; ++j) {
        const int n = (ngrp * 4 + j) * 16 + nloc;
        const float bv = bias[n];
#pragma unroll
        for (int r = 0; r < 8; ++r) {
            const int m = mtile * 16 + half * 8 + r;
            const float v = acc[j][r] + x0_t[(size_t)m * HID + n] + bv;
            const float h = tanhf(v);
            h_out[(size_t)m * HID + n]   = f2bf(h);
            h_final[(size_t)m * HID + n] = h;   // last step's write survives
        }
    }
}

// Layer 1: h1' = tanh(h1 @ wh1^T + h0' @ wx1^T + b1); also records bf16 h1' for logits.
__global__ __launch_bounds__(256) void rnn_step_l1(const unsigned short* __restrict__ h_in,
                                                   const unsigned short* __restrict__ h0_new,
                                                   const unsigned short* __restrict__ wh,
                                                   const unsigned short* __restrict__ wx,
                                                   const float* __restrict__ bias,
                                                   unsigned short* __restrict__ h_out,
                                                   unsigned short* __restrict__ h1_all_t,
                                                   float* __restrict__ h_final) {
    __shared__ __align__(16) unsigned short ldsB[2 * 4 * 16 * LROW];
    const int tid   = threadIdx.x;
    const int lane  = tid & 31;
    const int mtile = tid >> 5;
    const int ngrp  = blockIdx.x;
    const int nloc  = lane & 15;
    const int half  = lane >> 4;

    const v8f z = {0.f,0.f,0.f,0.f,0.f,0.f,0.f,0.f};
    v8f acc[4] = {z, z, z, z};

    // pass 1: h1 @ wh1^T
    accum_lds<4>(acc, ldsB,
                 h_in + ((size_t)mtile * 16 + nloc) * HID + half * 8,
                 wh   + (size_t)ngrp * 64 * HID,
                 HID, tid);
    // pass 2: h0' @ wx1^T (LDS reuse is safe: accum_lds ends with a barrier)
    accum_lds<4>(acc, ldsB,
                 h0_new + ((size_t)mtile * 16 + nloc) * HID + half * 8,
                 wx     + (size_t)ngrp * 64 * HID,
                 HID, tid);

#pragma unroll
    for (int j = 0; j < 4; ++j) {
        const int n = (ngrp * 4 + j) * 16 + nloc;
        const float bv = bias[n];
#pragma unroll
        for (int r = 0; r < 8; ++r) {
            const int m = mtile * 16 + half * 8 + r;
            const float h = tanhf(acc[j][r] + bv);
            const unsigned short hb = f2bf(h);
            h_out[(size_t)m * HID + n]    = hb;
            h1_all_t[(size_t)m * HID + n] = hb;
            h_final[(size_t)m * HID + n]  = h;
        }
    }
}

// ---------------- launch ----------------
extern "C" void kernel_launch(void* const* d_in, const int* in_sizes, int n_in,
                              void* d_out, int out_size, void* d_ws, size_t ws_size,
                              hipStream_t stream) {
    (void)in_sizes; (void)n_in; (void)out_size; (void)ws_size;

    const int*   ids  = (const int*)  d_in[0];   // [S,B]
    const float* hid0 = (const float*)d_in[1];   // [L,B,H]
    const float* emb  = (const float*)d_in[2];   // [V,E]
    const float* wx0  = (const float*)d_in[3];   // [H,E]
    const float* wx1  = (const float*)d_in[4];   // [1,H,H]
    const float* whw  = (const float*)d_in[5];   // [L,H,H]
    const float* whb  = (const float*)d_in[6];   // [L,H]
    const float* wyw  = (const float*)d_in[7];   // [V,H]
    const float* wyb  = (const float*)d_in[8];   // [V]
    float* out = (float*)d_out;

    // workspace carve-up
    char* ws = (char*)d_ws;
    size_t off = 0;
    auto take = [&](size_t bytes) -> char* {
        char* p = ws + off;
        off += (bytes + 255) & ~(size_t)255;
        return p;
    };
    unsigned short* Xe    = (unsigned short*)take((size_t)SB * EMB * 2);   // gathered embeds
    float*          X0    = (float*)         take((size_t)SB * HID * 4);   // embed @ wx0^T
    unsigned short* H1all = (unsigned short*)take((size_t)SB * HID * 2);   // h1 over time
    unsigned short* wx0b  = (unsigned short*)take((size_t)HID * EMB * 2);
    unsigned short* wx1b  = (unsigned short*)take((size_t)HID * HID * 2);
    unsigned short* wh0b  = (unsigned short*)take((size_t)HID * HID * 2);
    unsigned short* wh1b  = (unsigned short*)take((size_t)HID * HID * 2);
    unsigned short* wywb  = (unsigned short*)take((size_t)VOCAB * HID * 2);
    unsigned short* h0a   = (unsigned short*)take((size_t)BATCH * HID * 2);
    unsigned short* h0b   = (unsigned short*)take((size_t)BATCH * HID * 2);
    unsigned short* h1a   = (unsigned short*)take((size_t)BATCH * HID * 2);
    unsigned short* h1b   = (unsigned short*)take((size_t)BATCH * HID * 2);

    // 1) convert weights + initial hidden to bf16
    const int HH = HID * HID, HE = HID * EMB, VH = VOCAB * HID, BH = BATCH * HID;
    to_bf16_kernel<<<(HE + 255) / 256, 256, 0, stream>>>(wx0,       wx0b, HE);
    to_bf16_kernel<<<(HH + 255) / 256, 256, 0, stream>>>(wx1,       wx1b, HH);
    to_bf16_kernel<<<(HH + 255) / 256, 256, 0, stream>>>(whw,       wh0b, HH);
    to_bf16_kernel<<<(HH + 255) / 256, 256, 0, stream>>>(whw + HH,  wh1b, HH);
    to_bf16_kernel<<<(VH + 255) / 256, 256, 0, stream>>>(wyw,       wywb, VH);
    to_bf16_kernel<<<(BH + 255) / 256, 256, 0, stream>>>(hid0,      h0a,  BH);
    to_bf16_kernel<<<(BH + 255) / 256, 256, 0, stream>>>(hid0 + BH, h1a,  BH);

    // 2) gather embeddings (bf16)
    gather_embed_kernel<<<SB, 256, 0, stream>>>(ids, emb, Xe);

    // 3) hoisted input GEMM: X0 = Xe @ wx0^T  (M=16384, N=1024, K=1024), 64 N-tiles -> TN=4
    gemm_wmma_bf16<4, false><<<dim3(SB / 128, (HID / 16) / 4), 256, 0, stream>>>(
        Xe, wx0b, nullptr, X0, EMB, HID);

    // 4) sequential recurrence: two small WMMA GEMM kernels per step
    float* fin0 = out + (size_t)SB * VOCAB;       // final_h layer 0
    float* fin1 = fin0 + (size_t)BATCH * HID;     // final_h layer 1
    unsigned short *h0in = h0a, *h0out = h0b, *h1in = h1a, *h1out = h1b;
    for (int t = 0; t < SEQ; ++t) {
        rnn_step_l0<<<16, 256, 0, stream>>>(
            h0in, wh0b, X0 + (size_t)t * BATCH * HID, whb, h0out, fin0);
        rnn_step_l1<<<16, 256, 0, stream>>>(
            h1in, h0out, wh1b, wx1b, whb + HID, h1out,
            H1all + (size_t)t * BATCH * HID, fin1);
        unsigned short* s;
        s = h0in; h0in = h0out; h0out = s;
        s = h1in; h1in = h1out; h1out = s;
    }

    // 5) hoisted logits GEMM: out = H1all @ wy^T + wy_b
    //    (M=16384, N=10000, K=1024), 625 N-tiles = 5*125 -> TN=5, no tail, no branches
    gemm_wmma_bf16<5, true><<<dim3(SB / 128, (VOCAB / 16) / 5), 256, 0, stream>>>(
        H1all, wywb, wyb, out, HID, VOCAB);
}
